// PostProcess_56315611185544
// MI455X (gfx1250) — compile-verified, matrix operand order checked by reading, not run on hardware
//
#include <hip/hip_runtime.h>
#include <hip/hip_bf16.h>
#include <stdint.h>

#define BATCH 256
#define Q_ 900
#define C_ 91
#define QC_ (Q_ * C_)          // 81900
#define K_SEL 100
#define IOU_THR 0.7f
#define CONF_THR 0.3f
#define NBINS 8192             // 13-bit key prefix
#define CAND_MAX 1024
#define NPAD 112               // 7 * 16 tiles
#define OFF_SCORES 0
#define OFF_LABELS (BATCH * K_SEL)
#define OFF_BOXES  (2 * BATCH * K_SEL)
#define OFF_KEEP   (2 * BATCH * K_SEL + 4 * BATCH * K_SEL)

typedef __attribute__((ext_vector_type(2)))  float    v2f;
typedef __attribute__((ext_vector_type(8)))  float    v8f;
typedef __attribute__((ext_vector_type(16))) _Float16 v16h;

__device__ __forceinline__ uint32_t f2key(float f) {
    uint32_t u = __float_as_uint(f);
    return (u & 0x80000000u) ? ~u : (u | 0x80000000u);
}
__device__ __forceinline__ float key2f(uint32_t k) {
    uint32_t u = (k & 0x80000000u) ? (k ^ 0x80000000u) : ~k;
    return __uint_as_float(u);
}

// ---------------------------------------------------------------------------
// Kernel A: per-batch top-100 via histogram threshold + bitonic sort.
// One block per batch image; streams 81900 logits twice (bandwidth bound).
// ---------------------------------------------------------------------------
__global__ __launch_bounds__(256) void topk_kernel(
    const float* __restrict__ logits,
    const float* __restrict__ boxes_in,
    const float* __restrict__ tsizes,
    float* __restrict__ out)
{
    __shared__ unsigned int hist[NBINS];
    __shared__ unsigned int chunkSum[256];
    __shared__ unsigned int chunkPre[256];
    __shared__ unsigned long long cand[CAND_MAX];
    __shared__ unsigned int s_thrBin;
    __shared__ unsigned int s_candCount;

    const int b   = blockIdx.x;
    const int tid = threadIdx.x;
    const float* lp = logits + (size_t)b * QC_;

    for (int i = tid; i < NBINS; i += 256) hist[i] = 0u;
    if (tid == 0) s_candCount = 0u;
    __syncthreads();

    // Pass 1: histogram of 13-bit sortable-key prefixes
    for (int i = tid; i < QC_; i += 256) {
        __builtin_prefetch(lp + i + 4096, 0, 0);   // global_prefetch_b8
        uint32_t k = f2key(lp[i]);
        atomicAdd(&hist[k >> 19], 1u);
    }
    __syncthreads();

    // Descending-order chunk sums: chunk t covers orders [t*32, t*32+32), bin = NBINS-1-order
    {
        unsigned int s = 0u;
        const int o0 = tid * 32;
        for (int o = o0; o < o0 + 32; ++o) s += hist[NBINS - 1 - o];
        chunkSum[tid] = s;
    }
    __syncthreads();
    if (tid == 0) {
        unsigned int acc = 0u;
        for (int t = 0; t < 256; ++t) { chunkPre[t] = acc; acc += chunkSum[t]; }
    }
    __syncthreads();
    // Unique chunk where cumulative count crosses K_SEL finds the threshold bin.
    if (chunkPre[tid] < K_SEL && chunkPre[tid] + chunkSum[tid] >= K_SEL) {
        unsigned int acc = chunkPre[tid];
        const int o0 = tid * 32;
        for (int o = o0; o < o0 + 32; ++o) {
            unsigned int bin = NBINS - 1 - o;
            acc += hist[bin];
            if (acc >= K_SEL) { s_thrBin = bin; break; }
        }
    }
    __syncthreads();
    const uint32_t thrBin = s_thrBin;

    // Pass 2: collect all candidates with prefix >= threshold bin
    for (int i = tid; i < QC_; i += 256) {
        uint32_t k = f2key(lp[i]);
        if ((k >> 19) >= thrBin) {
            unsigned int slot = atomicAdd(&s_candCount, 1u);
            if (slot < CAND_MAX) {
                // ~idx in low word => equal keys sort with lowest index first (top_k tie rule)
                cand[slot] = ((unsigned long long)k << 32) |
                             (unsigned long long)(~(uint32_t)i);
            }
        }
    }
    __syncthreads();
    unsigned int cc = s_candCount; if (cc > CAND_MAX) cc = CAND_MAX;
    for (int i = tid; i < CAND_MAX; i += 256) if ((unsigned)i >= cc) cand[i] = 0ull;
    __syncthreads();

    // Bitonic sort, descending, N = 1024
    for (int kk = 2; kk <= CAND_MAX; kk <<= 1) {
        for (int jj = kk >> 1; jj > 0; jj >>= 1) {
            for (int i = tid; i < CAND_MAX; i += 256) {
                int ixj = i ^ jj;
                if (ixj > i) {
                    bool desc = ((i & kk) == 0);
                    unsigned long long a = cand[i], c = cand[ixj];
                    if (desc ? (a < c) : (a > c)) { cand[i] = c; cand[ixj] = a; }
                }
            }
            __syncthreads();
        }
    }

    // Emit top-100: scores (sigmoid), labels, scaled xyxy boxes
    if (tid < K_SEL) {
        unsigned long long e = cand[tid];
        uint32_t key = (uint32_t)(e >> 32);
        uint32_t idx = ~(uint32_t)e;
        float logit = key2f(key);
        float score = 1.0f / (1.0f + __expf(-logit));
        uint32_t q   = idx / C_;
        uint32_t lab = idx - q * C_;
        const float* bp = boxes_in + ((size_t)b * Q_ + q) * 4;
        float cx = bp[0], cy = bp[1], w = bp[2], h = bp[3];
        float img_h = tsizes[b * 2 + 0];
        float img_w = tsizes[b * 2 + 1];
        float x1 = (cx - 0.5f * w) * img_w;
        float y1 = (cy - 0.5f * h) * img_h;
        float x2 = (cx + 0.5f * w) * img_w;
        float y2 = (cy + 0.5f * h) * img_h;
        out[OFF_SCORES + b * K_SEL + tid] = score;
        out[OFF_LABELS + b * K_SEL + tid] = (float)lab;
        float* bx = out + OFF_BOXES + ((size_t)b * K_SEL + tid) * 4;
        bx[0] = x1; bx[1] = y1; bx[2] = x2; bx[3] = y2;
    }
}

// ---------------------------------------------------------------------------
// Kernel B: per-batch NMS. Pairwise union sums (area_i + area_j) computed as a
// rank-2 outer product on the WMMA unit (V_WMMA_F32_16X16X4_F32); intersection
// rectangles on VALU. Sequential greedy suppression, then keep = nms & conf.
// ---------------------------------------------------------------------------
__global__ __launch_bounds__(256) void nms_kernel(float* __restrict__ out)
{
    __shared__ float sb[NPAD * 4];
    __shared__ float sarea[NPAD];
    __shared__ float siou[NPAD * NPAD];   // ~49 KB
    __shared__ int   skeep[NPAD];

    const int b   = blockIdx.x;
    const int tid = threadIdx.x;
    const float* bx = out + OFF_BOXES + (size_t)b * K_SEL * 4;

    for (int i = tid; i < NPAD * 4; i += 256)
        sb[i] = (i < K_SEL * 4) ? bx[i] : 0.0f;   // pad with degenerate boxes
    __syncthreads();
    for (int i = tid; i < NPAD; i += 256) {
        float x1 = sb[i*4+0], y1 = sb[i*4+1], x2 = sb[i*4+2], y2 = sb[i*4+3];
        sarea[i] = (x2 - x1) * (y2 - y1);
        skeep[i] = 1;
    }
    __syncthreads();

    const int lane = tid & 31;
    const int wave = tid >> 5;

    // 7x7 tiles of 16x16 over the padded 112x112 IoU matrix, one wave per tile.
    for (int t = wave; t < 49; t += 8) {
        const int ti = t / 7, tj = t % 7;
        const bool lo = (lane < 16);
        const int l16 = lane & 15;
        float ai = sarea[ti * 16 + l16];
        float aj = sarea[tj * 16 + l16];
        v8f acc = {0.f, 0.f, 0.f, 0.f, 0.f, 0.f, 0.f, 0.f};

#if __has_builtin(__builtin_amdgcn_wmma_f32_16x16x4_f32)
        // A (16x4): col K0 = area_i, col K1 = ones, K2/K3 = 0
        // lanes 0-15 hold {K0,K1}, lanes 16-31 hold {K2,K3}
        v2f A; A.x = lo ? ai : 0.0f;  A.y = lo ? 1.0f : 0.0f;
        // B (4x16): row K0 = ones (lanes 0-15 of V0), row K1 = area_j (lanes 16-31 of V0)
        v2f Bm; Bm.x = lo ? 1.0f : aj; Bm.y = 0.0f;
        acc = __builtin_amdgcn_wmma_f32_16x16x4_f32(false, A, false, Bm,
                                                    (short)0, acc, false, false);
#else
        // Fallback: f16 16x16x32 WMMA (codegen-confirmed), areas pre-scaled by 2^-10
        v16h A = {}; v16h Bm = {};
        if (lo) { A[0] = (_Float16)(ai * 0.0009765625f); A[1] = (_Float16)1.0f; }
        Bm[0] = lo ? (_Float16)0.0009765625f : (_Float16)0.0f;
        Bm[1] = lo ? (_Float16)0.0f : (_Float16)(aj * 0.0009765625f * 0.0f + aj * 0.0009765625f);
        acc = __builtin_amdgcn_wmma_f32_16x16x32_f16(false, A, false, Bm,
                                                     (short)0, acc, false, false);
        #pragma unroll
        for (int r = 0; r < 8; ++r) acc[r] *= 1024.0f;
#endif
        // D layout: VGPR r holds element (M = r + (lane<16 ? 0 : 8), N = lane%16)
        const int N = lane & 15;
        const int mb = lo ? 0 : 8;
        #pragma unroll
        for (int r = 0; r < 8; ++r) {
            const int i = ti * 16 + mb + r;
            const int j = tj * 16 + N;
            float ax1 = sb[i*4+0], ay1 = sb[i*4+1], ax2 = sb[i*4+2], ay2 = sb[i*4+3];
            float bx1 = sb[j*4+0], by1 = sb[j*4+1], bx2 = sb[j*4+2], by2 = sb[j*4+3];
            float lx = fmaxf(ax1, bx1), ly = fmaxf(ay1, by1);
            float rx = fminf(ax2, bx2), ry = fminf(ay2, by2);
            float iw = fmaxf(rx - lx, 0.0f);
            float ih = fmaxf(ry - ly, 0.0f);
            float inter = iw * ih;
            float uni = acc[r] - inter;                 // area_i + area_j - inter
            siou[i * NPAD + j] = inter / fmaxf(uni, 1e-9f);
        }
    }
    __syncthreads();

    // Sequential greedy suppression (matches fori_loop semantics)
    for (int i = 0; i < K_SEL; ++i) {
        if (skeep[i]) {                                  // uniform across block
            for (int j = tid; j < K_SEL; j += 256)
                if (j > i && siou[i * NPAD + j] > IOU_THR) skeep[j] = 0;
        }
        __syncthreads();
    }

    if (tid < K_SEL) {
        float sc = out[OFF_SCORES + b * K_SEL + tid];
        out[OFF_KEEP + b * K_SEL + tid] =
            (skeep[tid] && sc > CONF_THR) ? 1.0f : 0.0f;
    }
}

extern "C" void kernel_launch(void* const* d_in, const int* in_sizes, int n_in,
                              void* d_out, int out_size, void* d_ws, size_t ws_size,
                              hipStream_t stream) {
    (void)in_sizes; (void)n_in; (void)out_size; (void)d_ws; (void)ws_size;
    const float* pred_logits  = (const float*)d_in[0];
    const float* pred_boxes   = (const float*)d_in[1];
    const float* target_sizes = (const float*)d_in[2];
    float* out = (float*)d_out;

    topk_kernel<<<BATCH, 256, 0, stream>>>(pred_logits, pred_boxes, target_sizes, out);
    nms_kernel<<<BATCH, 256, 0, stream>>>(out);
}
